// MultiAttn_34797825032550
// MI455X (gfx1250) — matrix operand order repaired and checked by reading, not compiled, not run on hardware
//
#include <hip/hip_runtime.h>

// MI455X (gfx1250) multi-head attention block:
//   cast f32->bf16, QKV GEMM (WMMA bf16, async-LDS double-buffered),
//   flash attention (WMMA bf16), FF GEMM. f32 accumulate everywhere.

typedef __attribute__((ext_vector_type(16))) __bf16 v16bf;
typedef __attribute__((ext_vector_type(8)))  __bf16 v8bf;
typedef __attribute__((ext_vector_type(8)))  float  v8f;
typedef int v4i_ __attribute__((vector_size(16)));   // matches builtin param type

#define D_MODEL 2048
#define NHEAD   16
#define HS      128
#define SEQ     2048
#define BATCH   4
#define MROWS   (BATCH * SEQ)     // 8192
#define NQKV    (3 * D_MODEL)     // 6144

#define AS1 __attribute__((address_space(1)))
#define AS3 __attribute__((address_space(3)))

#if defined(__AMDGCN__) && __has_builtin(__builtin_amdgcn_global_load_async_to_lds_b128)
#define HAS_ASYNC_LDS 1
#else
#define HAS_ASYNC_LDS 0
#endif

__device__ __forceinline__ v16bf make_frag(const __bf16* lo, const __bf16* hi) {
  v8bf a = *(const v8bf*)lo;
  v8bf b = *(const v8bf*)hi;
  return __builtin_shufflevector(a, b, 0,1,2,3,4,5,6,7,8,9,10,11,12,13,14,15);
}

// 16-byte global -> LDS copy: async DMA (ASYNCcnt) on gfx1250, sync fallback.
__device__ __forceinline__ void cp_async16(__bf16* lds_dst, const __bf16* g_src) {
#if HAS_ASYNC_LDS
  __builtin_amdgcn_global_load_async_to_lds_b128(
      (AS1 v4i_*)(void*)(g_src), (AS3 v4i_*)(void*)(lds_dst), 0, 0);
#else
  *(v8bf*)lds_dst = *(const v8bf*)g_src;
#endif
}

__device__ __forceinline__ void wait_async0() {
#if HAS_ASYNC_LDS
#if __has_builtin(__builtin_amdgcn_s_wait_asynccnt)
  __builtin_amdgcn_s_wait_asynccnt(0);
#else
  asm volatile("s_wait_asynccnt 0x0" ::: "memory");
#endif
#endif
}

// ---------------------------------------------------------------- cast kernel
__global__ void __launch_bounds__(256)
cast_f32_bf16(const float* __restrict__ in, __bf16* __restrict__ out, int n) {
  int i = (blockIdx.x * blockDim.x + threadIdx.x) * 8;
  if (i + 7 < n) {
    float4 a = *(const float4*)(in + i);
    float4 b = *(const float4*)(in + i + 4);
    v8bf o;
    o[0] = (__bf16)a.x; o[1] = (__bf16)a.y; o[2] = (__bf16)a.z; o[3] = (__bf16)a.w;
    o[4] = (__bf16)b.x; o[5] = (__bf16)b.y; o[6] = (__bf16)b.z; o[7] = (__bf16)b.w;
    *(v8bf*)(out + i) = o;
  } else {
    for (; i < n; ++i) out[i] = (__bf16)in[i];
  }
}

// ------------------------------------------------------- shared GEMM mainloop
// C[128x128] tile per workgroup (256 threads = 8 wave32), BK=64,
// double-buffered LDS fed by async global->LDS DMA.
// A: [M,K] bf16 row-major; Bw: [N,K] bf16 row-major (B^T storage).
#define BK 64

__device__ __forceinline__ void stage_tiles(
    const __bf16* __restrict__ A, const __bf16* __restrict__ Bw, int K,
    int mbase, int nbase, int kk, __bf16* ldsA, __bf16* ldsB, int tid) {
  const int row = tid >> 3;         // 0..31
  const int col = (tid & 7) * 8;    // 0,8,...,56
#pragma unroll
  for (int p = 0; p < 4; ++p) {
    const int r = row + p * 32;
    cp_async16(&ldsA[r * BK + col], &A[(size_t)(mbase + r) * K + kk + col]);
    cp_async16(&ldsB[r * BK + col], &Bw[(size_t)(nbase + r) * K + kk + col]);
  }
}

__device__ __forceinline__ void gemm_tile_mainloop(
    const __bf16* __restrict__ A, const __bf16* __restrict__ Bw, int K,
    int mbase, int nbase, __bf16 (*ldsA)[128 * BK], __bf16 (*ldsB)[128 * BK],
    v8f acc[8]) {
  const int tid  = threadIdx.x;
  const int wave = tid >> 5;
  const int lane = tid & 31;
  const int lh   = lane >> 4;   // lane half
  const int lr   = lane & 15;
  const int wm   = wave >> 2;   // 0..1 : 64 rows each
  const int wn   = wave & 3;    // 0..3 : 32 cols each

  v8f vz = {};
#pragma unroll
  for (int i = 0; i < 8; ++i) acc[i] = vz;

  const int nsteps = K / BK;
  stage_tiles(A, Bw, K, mbase, nbase, 0, ldsA[0], ldsB[0], tid);

#pragma unroll 1
  for (int i = 0; i < nsteps; ++i) {
    wait_async0();          // stage i landed (per-wave)
    __syncthreads();        // visible to all waves

    if (i + 1 < nsteps)     // overlap next-stage DMA with this stage's WMMAs
      stage_tiles(A, Bw, K, mbase, nbase, (i + 1) * BK,
                  ldsA[(i + 1) & 1], ldsB[(i + 1) & 1], tid);

    const __bf16* bufA = ldsA[i & 1];
    const __bf16* bufB = ldsB[i & 1];
#pragma unroll
    for (int ks = 0; ks < 2; ++ks) {
      v16bf bfrag[2];
#pragma unroll
      for (int ni = 0; ni < 2; ++ni) {
        // B fragment: lane = N column, elements = K[lh*8..+7], K[lh*8+16..+7]
        const __bf16* p = &bufB[(wn * 32 + ni * 16 + lr) * BK + ks * 32 + lh * 8];
        bfrag[ni] = make_frag(p, p + 16);
      }
#pragma unroll
      for (int mi = 0; mi < 4; ++mi) {
        // A fragment: lane = M row, elements = K[lh*8..+7], K[lh*8+16..+7]
        const __bf16* p = &bufA[(wm * 64 + mi * 16 + lr) * BK + ks * 32 + lh * 8];
        v16bf afrag = make_frag(p, p + 16);
#pragma unroll
        for (int ni = 0; ni < 2; ++ni) {
          acc[mi * 2 + ni] = __builtin_amdgcn_wmma_f32_16x16x32_bf16(
              false, afrag, false, bfrag[ni], (short)0, acc[mi * 2 + ni], false, false);
        }
      }
    }
    __syncthreads();        // all waves done reading buf (i&1) before reuse
  }
}

// ----------------------------------------------------------------- QKV GEMM
// out column e -> head h=e/384, which=(e%384)/128 (q/k/v), d=e%128.
__global__ void __launch_bounds__(256)
qkv_gemm_kernel(const __bf16* __restrict__ xh, const __bf16* __restrict__ wph,
                const float* __restrict__ bias,
                __bf16* __restrict__ qh, __bf16* __restrict__ kh,
                __bf16* __restrict__ vh, float* __restrict__ kv_out) {
  __shared__ __align__(16) __bf16 ldsA[2][128 * BK];
  __shared__ __align__(16) __bf16 ldsB[2][128 * BK];
  v8f acc[8];
  const int mbase = blockIdx.y * 128;
  const int nbase = blockIdx.x * 128;
  gemm_tile_mainloop(xh, wph, D_MODEL, mbase, nbase, ldsA, ldsB, acc);

  const int tid = threadIdx.x, wave = tid >> 5, lane = tid & 31;
  const int lh = lane >> 4, lr = lane & 15;
  const int wm = wave >> 2, wn = wave & 3;
#pragma unroll
  for (int mi = 0; mi < 4; ++mi) {
#pragma unroll
    for (int ni = 0; ni < 2; ++ni) {
      const int e = nbase + wn * 32 + ni * 16 + lr;
      const int h = e / 384;
      const int j = e - h * 384;
      const int which = j >> 7;   // uniform across the 16-col subtile
      const int d = j & 127;
      const float bv = bias[e];
#pragma unroll
      for (int r = 0; r < 8; ++r) {
        const int m = mbase + wm * 64 + mi * 16 + lh * 8 + r;  // token row
        const int b = m >> 11;
        const int s = m & 2047;
        const float val = acc[mi * 2 + ni][r] + bv;
        const size_t idx = ((size_t)(b * NHEAD + h) * SEQ + s) * HS + d;
        if (which == 0) {
          qh[idx] = (__bf16)val;
        } else if (which == 1) {
          kh[idx] = (__bf16)val;
          kv_out[idx] = val;                                    // kv[0] = k
        } else {
          vh[idx] = (__bf16)val;
          kv_out[(size_t)BATCH * NHEAD * SEQ * HS + idx] = val; // kv[1] = v
        }
      }
    }
  }
}

// ------------------------------------------------------------------ FF GEMM
__global__ void __launch_bounds__(256)
ff_gemm_kernel(const __bf16* __restrict__ aoh, const __bf16* __restrict__ wfh,
               const float* __restrict__ bias, float* __restrict__ out) {
  __shared__ __align__(16) __bf16 ldsA[2][128 * BK];
  __shared__ __align__(16) __bf16 ldsB[2][128 * BK];
  v8f acc[8];
  const int mbase = blockIdx.y * 128;
  const int nbase = blockIdx.x * 128;
  gemm_tile_mainloop(aoh, wfh, D_MODEL, mbase, nbase, ldsA, ldsB, acc);

  const int tid = threadIdx.x, wave = tid >> 5, lane = tid & 31;
  const int lh = lane >> 4, lr = lane & 15;
  const int wm = wave >> 2, wn = wave & 3;
#pragma unroll
  for (int mi = 0; mi < 4; ++mi) {
#pragma unroll
    for (int ni = 0; ni < 2; ++ni) {
      const int e = nbase + wn * 32 + ni * 16 + lr;
      const float bv = bias[e];
#pragma unroll
      for (int r = 0; r < 8; ++r) {
        const int m = mbase + wm * 64 + mi * 16 + lh * 8 + r;
        out[(size_t)m * D_MODEL + e] = acc[mi * 2 + ni][r] + bv;
      }
    }
  }
}

// ------------------------------------------------------- flash attention
// Block: 128 threads = 4 waves, each wave owns one 16-row query tile.
// Loop over 32-key steps; Q resident in VGPRs; V^T staged in LDS; P goes
// through LDS to convert C-layout -> A-layout.
__global__ void __launch_bounds__(128)
attn_kernel(const __bf16* __restrict__ qh, const __bf16* __restrict__ kh,
            const __bf16* __restrict__ vh, __bf16* __restrict__ aoh) {
  __shared__ __align__(16) __bf16 ldsVT[HS * 32];     // [d][key], 8 KB
  __shared__ __align__(16) __bf16 ldsP[4][16 * 32];   // per-wave probs, 4 KB

  const int tid  = threadIdx.x;
  const int wave = tid >> 5, lane = tid & 31;
  const int lh   = lane >> 4, lr = lane & 15;
  const int b = blockIdx.z, h = blockIdx.y;
  const int qbase = blockIdx.x * 64 + wave * 16;
  const size_t bh = (size_t)(b * NHEAD + h) * SEQ * HS;
  const __bf16* qp = qh + bh;
  const __bf16* kp = kh + bh;
  const __bf16* vp = vh + bh;

  // Q fragments: 4 K-chunks of 32 along hs=128
  v16bf qfrag[4];
#pragma unroll
  for (int c = 0; c < 4; ++c) {
    const __bf16* p = qp + (size_t)(qbase + lr) * HS + c * 32 + lh * 8;
    qfrag[c] = make_frag(p, p + 16);
  }

  v8f o[8];
  v8f vz = {};
#pragma unroll
  for (int i = 0; i < 8; ++i) o[i] = vz;
  float mrow[8], lrow[8];
#pragma unroll
  for (int r = 0; r < 8; ++r) { mrow[r] = -1e30f; lrow[r] = 0.f; }

  const float scale = 0.08838834764831845f;  // 1/sqrt(128)
  const int nkt = (blockIdx.x + 1) * 2;      // causal: keys < (block end)

#pragma unroll 1
  for (int kt = 0; kt < nkt; ++kt) {
    const int kb = kt * 32;
    // stage V^T [d][key]: 128-bit coalesced global reads, b16 LDS scatter
#pragma unroll
    for (int p = 0; p < 4; ++p) {
      const int key = p * 8 + (tid >> 4);        // 0..31
      const int d0  = (tid & 15) * 8;
      v8bf vv = *(const v8bf*)&vp[(size_t)(kb + key) * HS + d0];
#pragma unroll
      for (int j = 0; j < 8; ++j) ldsVT[(d0 + j) * 32 + key] = vv[j];
    }
    __syncthreads();

    if (kb <= qbase + 15) {  // wave-uniform: tile not fully masked
      // scores = Q @ K^T (16x32), accumulated over 4 hs-chunks
      v8f sc[2]; sc[0] = vz; sc[1] = vz;
#pragma unroll
      for (int c = 0; c < 4; ++c) {
#pragma unroll
        for (int ni = 0; ni < 2; ++ni) {
          const __bf16* p = kp + (size_t)(kb + ni * 16 + lr) * HS + c * 32 + lh * 8;
          v16bf bf = make_frag(p, p + 16);
          sc[ni] = __builtin_amdgcn_wmma_f32_16x16x32_bf16(
              false, qfrag[c], false, bf, (short)0, sc[ni], false, false);
        }
      }
      // online softmax (row stats per VGPR index, reduced across 16-lane half)
      float alpha[8];
#pragma unroll
      for (int r = 0; r < 8; ++r) {
        const int qrow = qbase + lh * 8 + r;
        float s0 = sc[0][r] * scale;
        float s1 = sc[1][r] * scale;
        if (kb + lr      > qrow) s0 = -65504.f;  // MIN_FP16 causal mask
        if (kb + 16 + lr > qrow) s1 = -65504.f;
        float t = fmaxf(s0, s1);
#pragma unroll
        for (int mk = 1; mk < 16; mk <<= 1) t = fmaxf(t, __shfl_xor(t, mk, 32));
        const float mnew = fmaxf(mrow[r], t);
        alpha[r] = __expf(mrow[r] - mnew);
        const float e0 = __expf(s0 - mnew);
        const float e1 = __expf(s1 - mnew);
        float sum = e0 + e1;
#pragma unroll
        for (int mk = 1; mk < 16; mk <<= 1) sum += __shfl_xor(sum, mk, 32);
        lrow[r] = lrow[r] * alpha[r] + sum;
        mrow[r] = mnew;
        // P to LDS in (row, col) layout for A-fragment reload
        const int row = lh * 8 + r;
        ldsP[wave][row * 32 + lr]      = (__bf16)e0;
        ldsP[wave][row * 32 + 16 + lr] = (__bf16)e1;
      }
#pragma unroll
      for (int t8 = 0; t8 < 8; ++t8)
#pragma unroll
        for (int r = 0; r < 8; ++r) o[t8][r] *= alpha[r];

      // P as A-fragment (16x32), V^T as B-fragments -> O += P @ V
      const __bf16* pp = &ldsP[wave][lr * 32 + lh * 8];
      v16bf pa = make_frag(pp, pp + 16);
#pragma unroll
      for (int nt = 0; nt < 8; ++nt) {
        const __bf16* vt = &ldsVT[(nt * 16 + lr) * 32 + lh * 8];
        v16bf vb = make_frag(vt, vt + 16);
        o[nt] = __builtin_amdgcn_wmma_f32_16x16x32_bf16(
            false, pa, false, vb, (short)0, o[nt], false, false);
      }
    }
    __syncthreads();
  }

  // finalize: O /= l, write attn output recombined to [B*S, D] bf16
  float rinv[8];
#pragma unroll
  for (int r = 0; r < 8; ++r) rinv[r] = 1.0f / lrow[r];
#pragma unroll
  for (int nt = 0; nt < 8; ++nt) {
#pragma unroll
    for (int r = 0; r < 8; ++r) {
      const int s = qbase + lh * 8 + r;
      const int d = nt * 16 + lr;
      aoh[((size_t)(b * SEQ + s)) * D_MODEL + h * HS + d] = (__bf16)(o[nt][r] * rinv[r]);
    }
  }
}

// ---------------------------------------------------------------- launcher
extern "C" void kernel_launch(void* const* d_in, const int* in_sizes, int n_in,
                              void* d_out, int out_size, void* d_ws, size_t ws_size,
                              hipStream_t stream) {
  const float* x      = (const float*)d_in[0];
  const float* w_proj = (const float*)d_in[1];
  const float* b_proj = (const float*)d_in[2];
  const float* w_ff   = (const float*)d_in[3];
  const float* b_ff   = (const float*)d_in[4];
  float* out = (float*)d_out;

  // workspace layout (bf16)
  __bf16* xh  = (__bf16*)d_ws;
  __bf16* wph = xh  + (size_t)MROWS * D_MODEL;
  __bf16* wfh = wph + (size_t)NQKV * D_MODEL;
  __bf16* qh  = wfh + (size_t)D_MODEL * D_MODEL;
  __bf16* kh  = qh  + (size_t)MROWS * D_MODEL;   // B*H*S*hs == B*S*D
  __bf16* vh  = kh  + (size_t)MROWS * D_MODEL;
  __bf16* aoh = vh  + (size_t)MROWS * D_MODEL;

  const int threads = 256;
  {
    size_t n = (size_t)MROWS * D_MODEL;
    cast_f32_bf16<<<(int)((n / 8 + threads - 1) / threads), threads, 0, stream>>>(x, xh, (int)n);
    n = (size_t)NQKV * D_MODEL;
    cast_f32_bf16<<<(int)((n / 8 + threads - 1) / threads), threads, 0, stream>>>(w_proj, wph, (int)n);
    n = (size_t)D_MODEL * D_MODEL;
    cast_f32_bf16<<<(int)((n / 8 + threads - 1) / threads), threads, 0, stream>>>(w_ff, wfh, (int)n);
  }

  float* kv_out = out + (size_t)MROWS * D_MODEL;  // next_prefix_kv [2,B,H,S,hs]

  qkv_gemm_kernel<<<dim3(NQKV / 128, MROWS / 128), 256, 0, stream>>>(
      xh, wph, b_proj, qh, kh, vh, kv_out);

  attn_kernel<<<dim3(SEQ / 64, NHEAD, BATCH), 128, 0, stream>>>(qh, kh, vh, aoh);

  ff_gemm_kernel<<<dim3(D_MODEL / 128, MROWS / 128), 256, 0, stream>>>(
      aoh, wfh, b_ff, out);
}
